// QSpatialCirConv_36051955482655
// MI455X (gfx1250) — 8-bit matrix operand order repaired and checked by reading, compiled, not run on hardware
//
#include <hip/hip_runtime.h>
#include <stdint.h>

typedef int v8i __attribute__((ext_vector_type(8)));
typedef int v4i_b __attribute__((vector_size(16)));

#define NB     8
#define NF     5
#define ZD     16
#define HH     48
#define WW     48
#define NTAP   27
#define NPIX   (ZD*HH*WW)
#define NBATCH 4
#define CH40   40
#define NELEM  (NBATCH*CH40*NPIX)

#define WS_BPACK   512
#define WS_WF      (512 + 40960)
#define WS_XQ      131072
#define XQ_PLANE   (NPIX*NB)

#if __has_builtin(__builtin_amdgcn_global_load_async_to_lds_b128)
#define USE_ASYNC_LDS 1
#else
#define USE_ASYNC_LDS 0
#endif

union V8 { v8i v; int e[8]; unsigned long long q[4]; };

__device__ __forceinline__ int q8clamp(float x, float inv) {
  float r = rintf(x * inv);
  r = fminf(127.f, fmaxf(-127.f, r));
  return (int)r;
}

#if USE_ASYNC_LDS
__device__ __forceinline__ void wait_async0() {
#if __has_builtin(__builtin_amdgcn_s_wait_asynccnt)
  __builtin_amdgcn_s_wait_asynccnt(0);
#else
  asm volatile("s_wait_asynccnt 0" ::: "memory");
#endif
}
#endif

__global__ void init_kernel(unsigned* hdr_u) {
  if (threadIdx.x < 2) hdr_u[threadIdx.x] = 0u;
}

__global__ __launch_bounds__(256) void maxred_kernel(const float* __restrict__ xr,
                                                     const float* __restrict__ xi,
                                                     unsigned* __restrict__ hdr_u) {
  __shared__ float sm[256];
  float mr = 0.f, mi = 0.f;
  for (size_t i = (size_t)blockIdx.x*256 + threadIdx.x; i < (size_t)NELEM;
       i += (size_t)gridDim.x*256) {
    mr = fmaxf(mr, fabsf(xr[i]));
    mi = fmaxf(mi, fabsf(xi[i]));
  }
  sm[threadIdx.x] = mr; __syncthreads();
  for (int s = 128; s > 0; s >>= 1) {
    if (threadIdx.x < s) sm[threadIdx.x] = fmaxf(sm[threadIdx.x], sm[threadIdx.x + s]);
    __syncthreads();
  }
  if (threadIdx.x == 0) atomicMax(&hdr_u[0], __float_as_uint(sm[0]));
  __syncthreads();
  sm[threadIdx.x] = mi; __syncthreads();
  for (int s = 128; s > 0; s >>= 1) {
    if (threadIdx.x < s) sm[threadIdx.x] = fmaxf(sm[threadIdx.x], sm[threadIdx.x + s]);
    __syncthreads();
  }
  if (threadIdx.x == 0) atomicMax(&hdr_u[1], __float_as_uint(sm[0]));
}

__global__ __launch_bounds__(256) void wdft_kernel(const float* __restrict__ wt,
                                                   const float* __restrict__ bias,
                                                   float* __restrict__ hdrf,
                                                   float* __restrict__ wfre,
                                                   float* __restrict__ wfim,
                                                   char*  __restrict__ bpack) {
  __shared__ float red[256];
  __shared__ float sc[2];
  const float C8[8] = {1.f, 0.70710678f, 0.f, -0.70710678f, -1.f, -0.70710678f, 0.f, 0.70710678f};
  const float S8[8] = {0.f, 0.70710678f, 1.f, 0.70710678f, 0.f, -0.70710678f, -1.f, -0.70710678f};
  const int tid = threadIdx.x;

  float mr = 0.f, mi = 0.f;
  for (int u = tid; u < 8640; u += 256) {
    int t  = u % 27;
    int kf = (u / 27) % 5;
    int ib = (u / 135) % 8;
    int ob = u / 1080;
    float re = 0.f, im = 0.f;
    #pragma unroll
    for (int n = 0; n < 8; ++n) {
      float wv = wt[((ob*8 + ib)*8 + n)*27 + t];
      int a = (kf * n) & 7;
      re += wv * C8[a];
      im -= wv * S8[a];
    }
    wfre[u] = re; wfim[u] = im;
    mr = fmaxf(mr, fabsf(re)); mi = fmaxf(mi, fabsf(im));
  }
  red[tid] = mr; __syncthreads();
  for (int s = 128; s > 0; s >>= 1) { if (tid < s) red[tid] = fmaxf(red[tid], red[tid+s]); __syncthreads(); }
  float maxr = red[0]; __syncthreads();
  red[tid] = mi; __syncthreads();
  for (int s = 128; s > 0; s >>= 1) { if (tid < s) red[tid] = fmaxf(red[tid], red[tid+s]); __syncthreads(); }
  float maxi = red[0]; __syncthreads();
  if (tid == 0) {
    sc[0] = maxr * (1.f/127.f);  sc[1] = maxi * (1.f/127.f);
    hdrf[2] = sc[0];  hdrf[3] = sc[1];
  }
  if (tid < 40) {
    int kf = tid % 5, ob = tid / 5;
    float re = 0.f, im = 0.f;
    #pragma unroll
    for (int c = 0; c < 8; ++c) {
      float bv = bias[ob*8 + c];
      int a = (kf * c) & 7;
      re += bv * C8[a];
      im -= bv * S8[a];
    }
    hdrf[16 + tid*2]     = re;
    hdrf[16 + tid*2 + 1] = im;
  }
  __syncthreads();

  float swr = sc[0], swi = sc[1];
  float invr = swr > 0.f ? 1.f/swr : 0.f;
  float invi = swi > 0.f ? 1.f/swi : 0.f;
  for (int q = tid; q < 10240; q += 256) {
    int v    = q & 7;
    int lane = (q >> 3) & 31;
    int s    = (q >> 8) & 3;
    int kf   = (q >> 10) % 5;
    int g    = q / 5120;
    int n    = lane & 15;
    int ob   = (n < 8) ? n : n - 8;
    unsigned dw = 0;
    #pragma unroll
    for (int bb = 0; bb < 4; ++bb) {
      int K = s*64 + (v >> 1)*16 + (lane >> 4)*8 + (v & 1)*4 + bb;
      int t = K >> 3, c = K & 7;
      int val = 0;
      if (t < NTAP) {
        int wix = ((ob*8 + c)*5 + kf)*27 + t;
        bool realsel = (g == 0) ? (n < 8) : (n >= 8);
        float src = realsel ? wfre[wix] : wfim[wix];
        float inv = realsel ? invr : invi;
        val = q8clamp(src, inv);
      }
      dw |= ((unsigned)(val & 0xff)) << (8*bb);
    }
    *(unsigned*)(bpack + (size_t)q*4) = dw;
  }
}

__global__ __launch_bounds__(256) void quantx_kernel(const float* __restrict__ xr,
                                                     const float* __restrict__ xi,
                                                     const float* __restrict__ hdrf,
                                                     char* __restrict__ xq) {
  int gid = blockIdx.x*256 + threadIdx.x;
  int x  = gid % WW;  int t = gid / WW;
  int y  = t % HH;    t /= HH;
  int z  = t % ZD;    t /= ZD;
  int kf = t % NF;    int b = t / NF;

  const unsigned* hu = (const unsigned*)hdrf;
  float maxr = __uint_as_float(hu[0]);
  float maxi = __uint_as_float(hu[1]);
  float invr = maxr > 0.f ? 127.f/maxr : 0.f;
  float invi = maxi > 0.f ? 127.f/maxi : 0.f;
  bool killim = (kf == 0) || (kf == 4);

  unsigned long long qr = 0ull, qi = 0ull;
  #pragma unroll
  for (int ib = 0; ib < 8; ++ib) {
    size_t idx = ((size_t)((b*CH40 + ib*NF + kf)*ZD + z)*HH + y)*WW + x;
    unsigned vr = (unsigned)(q8clamp(xr[idx], invr) & 0xff);
    unsigned vi = killim ? 0u : (unsigned)(q8clamp(xi[idx], invi) & 0xff);
    qr |= (unsigned long long)vr << (8*ib);
    qi |= (unsigned long long)vi << (8*ib);
  }
  char* base = xq + (size_t)((b*NF + kf)*2) * XQ_PLANE + (size_t)((z*HH + y)*WW + x) * NB;
  *(unsigned long long*)(base)            = qr;
  *(unsigned long long*)(base + XQ_PLANE) = qi;
}

__global__ __launch_bounds__(256) void conv_wmma_kernel(const char* __restrict__ xq,
                                                        const char* __restrict__ bpack,
                                                        const float* __restrict__ hdrf,
                                                        float2* __restrict__ out) {
  __shared__ v8i bshv[256];
  __shared__ float tile[8][256];
  const int tid  = threadIdx.x;
  const int w    = tid >> 5, lane = tid & 31;
  const int m    = lane & 15, half = lane >> 4;

  const int y   = blockIdx.x * 8 + w;
  const int zxt = blockIdx.y;
  const int xt  = zxt % 3, z = zxt / 3;
  const int bkf = blockIdx.z;
  const int kf  = bkf % 5, b = bkf / 5;
  const int x0  = xt * 16;

  {
    char* ldst = (char*)bshv;
#if USE_ASYNC_LDS
    #pragma unroll
    for (int r = 0; r < 2; ++r) {
      const char* gsrc = bpack + (size_t)(r*5 + kf)*4096 + (size_t)tid*16;
      __builtin_amdgcn_global_load_async_to_lds_b128(
          (__attribute__((address_space(1))) v4i_b*)(gsrc),
          (__attribute__((address_space(3))) v4i_b*)(ldst + r*4096 + tid*16),
          0, 0);
    }
    wait_async0();
#else
    #pragma unroll
    for (int r = 0; r < 2; ++r) {
      const int4* gsrc = (const int4*)(bpack + (size_t)(r*5 + kf)*4096) + tid;
      ((int4*)(ldst + r*4096))[tid] = *gsrc;
    }
#endif
  }
  __syncthreads();

  const char* baseR = xq + (size_t)((b*NF + kf)*2) * XQ_PLANE;
  const char* baseI = baseR + XQ_PLANE;
  {
    int off = ((z*HH + y)*WW + x0) * NB;
    __builtin_prefetch(baseR + off, 0, 1);
    __builtin_prefetch(baseI + off, 0, 1);
  }

  v8i acc1 = {0,0,0,0,0,0,0,0};
  v8i acc2 = {0,0,0,0,0,0,0,0};
  const int xxm = x0 + m - 1;

  #pragma unroll
  for (int s = 0; s < 4; ++s) {
    V8 a1, a2, b1, b2;
    b1.v = bshv[s*32 + lane];
    b2.v = bshv[128 + s*32 + lane];
    #pragma unroll
    for (int j = 0; j < 4; ++j) {
      int t = s*8 + j*2 + half;
      unsigned long long vr = 0ull, vi = 0ull;
      if (t < NTAP) {
        int dz = t / 9, rr = t % 9;
        int zz = z + dz - 1;
        int yy = y + rr/3 - 1;
        int xx = xxm + rr%3;
        if ((unsigned)zz < ZD && (unsigned)yy < HH && (unsigned)xx < WW) {
          size_t off = (size_t)((zz*HH + yy)*WW + xx) * NB;
          vr = *(const unsigned long long*)(baseR + off);
          vi = *(const unsigned long long*)(baseI + off);
        }
      }
      a1.q[j] = vr;
      a2.q[j] = vi;
    }
    acc1 = __builtin_amdgcn_wmma_i32_16x16x64_iu8(true, a1.v, true, b1.v, acc1, false, false);
    acc2 = __builtin_amdgcn_wmma_i32_16x16x64_iu8(true, a2.v, true, b2.v, acc2, false, false);
  }

  const unsigned* hu = (const unsigned*)hdrf;
  float sar = __uint_as_float(hu[0]) * (1.f/127.f);
  float sai = __uint_as_float(hu[1]) * (1.f/127.f);
  float swr = hdrf[2], swi = hdrf[3];
  float crr = sar*swr, cii = sai*swi, cri = sar*swi, cir = sai*swr;
  const float* bd = hdrf + 16;
  int n   = lane & 15;
  int obn = (n < 8) ? n : n - 8;
  float badd = (n < 8) ? bd[(obn*NF + kf)*2] : bd[(obn*NF + kf)*2 + 1];

  V8 r1, r2; r1.v = acc1; r2.v = acc2;
  #pragma unroll
  for (int r = 0; r < 8; ++r) {
    int mm = r + half*8;
    float p1 = (float)r1.e[r], p2 = (float)r2.e[r];
    float val = (n < 8) ? (crr*p1 - cii*p2) : (cri*p1 + cir*p2);
    tile[w][mm*16 + n] = val + badd;
  }
  __builtin_amdgcn_wave_barrier();

  #pragma unroll
  for (int i = 0; i < 4; ++i) {
    int ob = i*2 + half;
    float2 v;
    v.x = tile[w][m*16 + ob];
    v.y = tile[w][m*16 + 8 + ob];
    size_t oidx = ((size_t)((b*CH40 + ob*NF + kf)*ZD + z)*HH + y)*WW + (x0 + m);
    out[oidx] = v;
  }
}

extern "C" void kernel_launch(void* const* d_in, const int* in_sizes, int n_in,
                              void* d_out, int out_size, void* d_ws, size_t ws_size,
                              hipStream_t stream) {
  const float* xr   = (const float*)d_in[0];
  const float* xi   = (const float*)d_in[1];
  const float* wt   = (const float*)d_in[2];
  const float* bias = (const float*)d_in[3];

  char*     ws    = (char*)d_ws;
  unsigned* hdr_u = (unsigned*)ws;
  float*    hdrf  = (float*)ws;
  char*     bpack = ws + WS_BPACK;
  float*    wfre  = (float*)(ws + WS_WF);
  float*    wfim  = wfre + 8640;
  char*     xq    = ws + WS_XQ;
  float2*   out   = (float2*)d_out;

  init_kernel   <<<1,    32,  0, stream>>>(hdr_u);
  maxred_kernel <<<2880, 256, 0, stream>>>(xr, xi, hdr_u);
  wdft_kernel   <<<1,    256, 0, stream>>>(wt, bias, hdrf, wfre, wfim, bpack);
  quantx_kernel <<<2880, 256, 0, stream>>>(xr, xi, hdrf, xq);
  conv_wmma_kernel<<<dim3(6, 48, 20), 256, 0, stream>>>(xq, bpack, hdrf, out);

  (void)in_sizes; (void)n_in; (void)out_size; (void)ws_size;
}
